// PFA_LLAMA_64020782514427
// MI455X (gfx1250) — compile-verified, matrix operand order checked by reading, not run on hardware
//
#include <hip/hip_runtime.h>
#include <hip/hip_bf16.h>

// ---------------------------------------------------------------------------
// Types for CDNA5 WMMA (wave32): 16x16x32 bf16 -> f32
// ---------------------------------------------------------------------------
typedef __attribute__((ext_vector_type(16))) __bf16 v16bf;
typedef __attribute__((ext_vector_type(8)))  float  v8f;
typedef __attribute__((ext_vector_type(4)))  unsigned int u32x4;
typedef __attribute__((ext_vector_type(8)))  int i32x8;
typedef __attribute__((ext_vector_type(4)))  int i32x4;

union Frag {
  v16bf v;
  uint4 u[2];
  __bf16 h[16];
};

__device__ inline v8f wmma_bf16(v16bf a, v16bf b, v8f c) {
  // (neg_a, A, neg_b, B, c_mod, C, reuse_a, reuse_b)
  return __builtin_amdgcn_wmma_f32_16x16x32_bf16(false, a, false, b, (short)0, c,
                                                 false, false);
}

// ---------------------------------------------------------------------------
// CDNA5 async global->LDS (ASYNCcnt) helpers.
// VDST = VGPR with LDS byte address; VADDR = 64-bit global address (GV mode).
// Flat pointers to LDS carry the wave-relative LDS offset in bits [31:0].
// ---------------------------------------------------------------------------
__device__ inline unsigned lds_addr32(const void* p) {
  return (unsigned)(size_t)p;
}

__device__ inline void async_load_b128(unsigned lds_off, const void* gaddr) {
  asm volatile("global_load_async_to_lds_b128 %0, %1, off"
               :
               : "v"(lds_off), "v"(gaddr)
               : "memory");
}

template <int N>
__device__ inline void wait_asynccnt() {
  asm volatile("s_wait_asynccnt %0" : : "i"(N) : "memory");
}

// ---------------------------------------------------------------------------
// CDNA5 Tensor Data Mover: 2D weight tile (128 rows x 32 bf16, row stride K
// elements) -> LDS with hardware padding so rows land at a 40-half stride
// (pad_interval: 64B = one row; pad_amount: 16B). Descriptor per ISA 8.3/8.4.
// ---------------------------------------------------------------------------
#if __has_builtin(__builtin_amdgcn_tensor_load_to_lds)
#define HAVE_TDM 1
#else
#define HAVE_TDM 0
#endif

#if HAVE_TDM
__device__ inline void tdm_load_tile_128x32(unsigned lds, const void* gsrc,
                                            int strideK) {
  unsigned long long ga = (unsigned long long)(size_t)gsrc;
  u32x4 g0;
  g0[0] = 1u;             // count = 1 (valid descriptor)
  g0[1] = lds;            // lds_addr (bytes)
  g0[2] = (unsigned)ga;   // global_addr[31:0]
  g0[3] = (unsigned)((ga >> 32) & 0x1FFFFFFu) | 0x80000000u;  // addr[56:32]|type=2
  i32x8 g1;
  // data_size=2B (1), pad_enable, pad_interval=3 (64B), pad_amount=3 (4 DWORDs)
  g1[0] = (int)((1u << 16) | (1u << 20) | (3u << 22) | (3u << 25));
  g1[1] = (int)(32u << 16);   // tensor_dim0 = 32 (elements per tile row)
  g1[2] = (int)(128u << 16);  // tensor_dim1 = 128
  g1[3] = (int)(32u << 16);   // tile_dim0 = 32
  g1[4] = 128;                // tile_dim1 = 128, tile_dim2 = 0
  g1[5] = strideK;            // tensor_dim0_stride (elements)
  g1[6] = 0;
  g1[7] = 0;
  i32x4 gz = {};
#if __has_include(<hip/amd_detail/amd_gfx1250_TDM.h>)
  i32x8 z8 = {};
  __builtin_amdgcn_tensor_load_to_lds(g0, g1, gz, gz, z8, 0);
#else
  __builtin_amdgcn_tensor_load_to_lds(g0, g1, gz, gz, 0);
#endif
}
#endif

template <int N>
__device__ inline void wait_tensorcnt() {
#if __has_builtin(__builtin_amdgcn_s_wait_tensorcnt)
  __builtin_amdgcn_s_wait_tensorcnt((short)N);
#else
  asm volatile("s_wait_tensorcnt %0" : : "i"(N) : "memory");
#endif
}

__device__ __constant__ float NF4_CODES_C[16] = {
    -1.0f, -0.6961928009986877f, -0.5250730514526367f, -0.39491748809814453f,
    -0.28444138169288635f, -0.18477343022823334f, -0.09105003625154495f, 0.0f,
    0.07958029955625534f, 0.16093020141124725f, 0.24611230194568634f,
    0.33791524171829224f, 0.44070982933044434f, 0.5626170039176941f,
    0.7229568362236023f, 1.0f};

// ---------------------------------------------------------------------------
// NF4 fake quant->dequant: per-64 block absmax, nearest code, emit bf16.
// ---------------------------------------------------------------------------
__global__ __launch_bounds__(256) void nf4_dequant_kernel(
    const float* __restrict__ w, __bf16* __restrict__ out, int nblocks) {
  int blk = blockIdx.x * blockDim.x + threadIdx.x;
  if (blk >= nblocks) return;
  const float* p = w + (size_t)blk * 64;
  float amax = 0.0f;
  for (int i = 0; i < 64; i += 4) {
    float4 v = *(const float4*)(p + i);
    amax = fmaxf(amax, fmaxf(fmaxf(fabsf(v.x), fabsf(v.y)),
                             fmaxf(fabsf(v.z), fabsf(v.w))));
  }
  float inv = 1.0f / fmaxf(amax, 1e-12f);
  __bf16* o = out + (size_t)blk * 64;
  for (int i = 0; i < 64; ++i) {
    float nx = p[i] * inv;
    int idx = 0;
#pragma unroll
    for (int j = 0; j < 15; ++j)
      idx += (nx > 0.5f * (NF4_CODES_C[j] + NF4_CODES_C[j + 1])) ? 1 : 0;
    o[i] = (__bf16)(NF4_CODES_C[idx] * amax);
  }
}

// ---------------------------------------------------------------------------
// RMSNorm: one workgroup per row. OutT = __bf16 (internal) or float (final).
// ---------------------------------------------------------------------------
template <typename OutT>
__global__ __launch_bounds__(256) void rmsnorm_kernel(
    const float* __restrict__ x, const float* __restrict__ w,
    OutT* __restrict__ out, int D) {
  const int row = blockIdx.x;
  const int tid = threadIdx.x;
  const float* xr = x + (size_t)row * D;
  float s = 0.0f;
  for (int i = tid; i < D; i += 256) {
    float v = xr[i];
    s += v * v;
  }
  __shared__ float red[256];
  red[tid] = s;
  __syncthreads();
  for (int o = 128; o > 0; o >>= 1) {
    if (tid < o) red[tid] += red[tid + o];
    __syncthreads();
  }
  float inv = rsqrtf(red[0] / (float)D + 1e-5f);
  OutT* orow = out + (size_t)row * D;
  for (int i = tid; i < D; i += 256) orow[i] = (OutT)(xr[i] * inv * w[i]);
}

// ---------------------------------------------------------------------------
// C[M,N] = A[M,K] * Bw[N,K]^T, bf16 WMMA, f32 out. Tile 128x128x32.
// Double-buffered LDS: A tile via per-lane GLOBAL_LOAD_ASYNC_TO_LDS_B128
// (ASYNCcnt), B (weight) tile via one TDM TENSOR_LOAD_TO_LDS descriptor
// issued by wave 0 (TENSORcnt), with hardware LDS padding matching the
// 40-half row stride. Tile k+1 stays in flight while tile k is consumed.
// 256 threads = 8 waves; wave (wm,wn) owns 32x64 = 2x4 subtiles of 16x16.
// ---------------------------------------------------------------------------
__global__ __launch_bounds__(256) void gemm_bf16_wmma_kernel(
    const __bf16* __restrict__ A, const __bf16* __restrict__ Bw,
    float* __restrict__ C, int M, int N, int K) {
  __shared__ __align__(16) __bf16 sA[2][128][40];
  __shared__ __align__(16) __bf16 sB[2][128][40];
  const int tid = threadIdx.x;
  const int lane = tid & 31;
  const int wave = tid >> 5;
  const int bm = blockIdx.x * 128;
  const int bn = blockIdx.y * 128;
  const int wm = (wave & 3) * 32;
  const int wn = (wave >> 2) * 64;
  const int ldr = tid >> 1;        // tile row this thread loads
  const int ldc = (tid & 1) * 16;  // half-row (16 halves = 32B)
  const int nrow = lane & 15;
  const int khalf = (lane >> 4) * 16;

  v8f acc[2][4] = {};

  const __bf16* gA = A + (size_t)(bm + ldr) * K + ldc;
  const __bf16* gBt = Bw + (size_t)bn * K;           // B tile origin (TDM)
  const __bf16* gB = Bw + (size_t)(bn + ldr) * K + ldc;  // per-lane fallback

  unsigned lA[2], lB[2], lBt[2];
#pragma unroll
  for (int bufi = 0; bufi < 2; ++bufi) {
    lA[bufi] = lds_addr32(&sA[bufi][ldr][ldc]);
    lB[bufi] = lds_addr32(&sB[bufi][ldr][ldc]);
    lBt[bufi] = lds_addr32(&sB[bufi][0][0]);
  }

  const int nk = K >> 5;

  auto issue_tile = [&](int kt, int bufi) {
    const __bf16* nA = gA + kt * 32;
    async_load_b128(lA[bufi], nA);
    async_load_b128(lA[bufi] + 16, nA + 8);
#if HAVE_TDM
    if (wave == 0) tdm_load_tile_128x32(lBt[bufi], gBt + kt * 32, K);
#else
    const __bf16* nB = gB + kt * 32;
    async_load_b128(lB[bufi], nB);
    async_load_b128(lB[bufi] + 16, nB + 8);
#endif
  };
#if HAVE_TDM
  constexpr int ASYNC_AHEAD = 2;  // A-only async ops per tile
#else
  constexpr int ASYNC_AHEAD = 4;
#endif

  issue_tile(0, 0);

  for (int kt = 0; kt < nk; ++kt) {
    const int cur = kt & 1;
    if (kt + 1 < nk) {
      if (kt + 2 < nk) {
        __builtin_prefetch(gA + (kt + 2) * 32, 0, 1);  // global_prefetch_b8
        __builtin_prefetch(gBt + (kt + 2) * 32, 0, 1);
      }
      issue_tile(kt + 1, cur ^ 1);
      wait_asynccnt<ASYNC_AHEAD>();  // tile kt landed; kt+1 stays in flight
#if HAVE_TDM
      wait_tensorcnt<1>();
#endif
    } else {
      wait_asynccnt<0>();
#if HAVE_TDM
      wait_tensorcnt<0>();
#endif
    }
    __syncthreads();  // all waves' tile-kt DMA writes visible

    Frag fa[2], fb[4];
#pragma unroll
    for (int i = 0; i < 2; ++i) {
      const __bf16* sp = &sA[cur][wm + 16 * i + nrow][khalf];
      fa[i].u[0] = *(const uint4*)sp;
      fa[i].u[1] = *(const uint4*)(sp + 8);
    }
#pragma unroll
    for (int j = 0; j < 4; ++j) {
      const __bf16* sp = &sB[cur][wn + 16 * j + nrow][khalf];
      fb[j].u[0] = *(const uint4*)sp;
      fb[j].u[1] = *(const uint4*)(sp + 8);
    }
#pragma unroll
    for (int i = 0; i < 2; ++i)
#pragma unroll
      for (int j = 0; j < 4; ++j)
        acc[i][j] = wmma_bf16(fa[i].v, fb[j].v, acc[i][j]);

    __syncthreads();  // everyone done reading buf[cur] before it is refilled
  }

  const int ncol = lane & 15;
  const int moff = (lane >> 4) * 8;
#pragma unroll
  for (int i = 0; i < 2; ++i)
#pragma unroll
    for (int j = 0; j < 4; ++j)
#pragma unroll
      for (int e = 0; e < 8; ++e) {
        int m = bm + wm + 16 * i + e + moff;
        int n = bn + wn + 16 * j + ncol;
        C[(size_t)m * N + n] = acc[i][j][e];
      }
}

// ---------------------------------------------------------------------------
// LoRA: t[M,R] = h(bf16) @ a^T ; then y += scale * t @ b^T
// ---------------------------------------------------------------------------
__global__ __launch_bounds__(256) void lora_t_kernel(
    const __bf16* __restrict__ h, const float* __restrict__ a,
    float* __restrict__ t, int M, int R, int D) {
  int w = blockIdx.x * 8 + (threadIdx.x >> 5);
  int lane = threadIdx.x & 31;
  if (w >= M * R) return;
  int m = w / R, r = w % R;
  float s = 0.0f;
  for (int d = lane; d < D; d += 32)
    s += (float)h[(size_t)m * D + d] * a[(size_t)r * D + d];
#pragma unroll
  for (int o = 16; o > 0; o >>= 1) s += __shfl_xor(s, o, 32);
  if (lane == 0) t[w] = s;
}

__global__ __launch_bounds__(256) void lora_add_kernel(
    float* __restrict__ y, const float* __restrict__ t,
    const float* __restrict__ bmat, int M, int N, int R, float scale) {
  size_t i = (size_t)blockIdx.x * blockDim.x + threadIdx.x;
  if (i >= (size_t)M * N) return;
  int m = (int)(i / N), n = (int)(i % N);
  float s = 0.0f;
#pragma unroll
  for (int r = 0; r < 8; ++r) s += t[(size_t)m * R + r] * bmat[(size_t)n * R + r];
  y[i] += scale * s;
}

// ---------------------------------------------------------------------------
// RoPE (f32 in, bf16 out), layout [B*S, H, 128]
// ---------------------------------------------------------------------------
__global__ __launch_bounds__(256) void rope_to_bf16_kernel(
    const float* __restrict__ y, __bf16* __restrict__ o, int Mtot, int S, int H) {
  const int half = 64, Dh = 128;
  size_t i = (size_t)blockIdx.x * blockDim.x + threadIdx.x;
  size_t tot = (size_t)Mtot * H * half;
  if (i >= tot) return;
  int hi = (int)(i % half);
  size_t r2 = i / half;
  int h = (int)(r2 % H);
  size_t m = r2 / H;
  int s = (int)(m % S);
  float texp = (float)(2 * hi) * (1.0f / 128.0f);
  float invf = __expf(-texp * 9.210340371976184f);  // 10000^-texp
  float ang = (float)s * invf;
  float sn, cs;
  __sincosf(ang, &sn, &cs);
  size_t base = m * (size_t)(H * Dh) + (size_t)h * Dh;
  float x1 = y[base + hi];
  float x2 = y[base + hi + half];
  o[base + hi] = (__bf16)(x1 * cs - x2 * sn);
  o[base + hi + half] = (__bf16)(x2 * cs + x1 * sn);
}

// ---------------------------------------------------------------------------
// Elementwise helpers
// ---------------------------------------------------------------------------
__global__ __launch_bounds__(256) void tobf16_kernel(const float* __restrict__ in,
                                                     __bf16* __restrict__ out,
                                                     size_t n) {
  size_t i = (size_t)blockIdx.x * blockDim.x + threadIdx.x;
  if (i < n) out[i] = (__bf16)in[i];
}

__global__ __launch_bounds__(256) void silu_bf16_kernel(const float* __restrict__ y,
                                                        __bf16* __restrict__ g,
                                                        size_t n) {
  size_t i = (size_t)blockIdx.x * blockDim.x + threadIdx.x;
  if (i < n) {
    float v = y[i];
    g[i] = (__bf16)(v / (1.0f + __expf(-v)));
  }
}

__global__ __launch_bounds__(256) void mul_bf16_kernel(const float* __restrict__ up,
                                                       __bf16* __restrict__ g,
                                                       size_t n) {
  size_t i = (size_t)blockIdx.x * blockDim.x + threadIdx.x;
  if (i < n) g[i] = (__bf16)(up[i] * (float)g[i]);
}

__global__ __launch_bounds__(256) void add_residual_kernel(float* __restrict__ x,
                                                           const float* __restrict__ y,
                                                           size_t n) {
  size_t i = (size_t)blockIdx.x * blockDim.x + threadIdx.x;
  if (i < n) x[i] += y[i];
}

// ---------------------------------------------------------------------------
// Flash attention, 1 wave per (b, h, 16-row q block). Dh = 128, causal + pad.
// Scores via 4x WMMA bf16 (K-dim = Dh); V tile is brought into LDS with async
// loads issued BEFORE the softmax VALU work so ASYNCcnt hides the fill; PV via
// 8x WMMA per 32-k step. Online-softmax stats match the C-matrix VGPR layout
// (row = e + 8*(lane>=16)).
// ---------------------------------------------------------------------------
__global__ __launch_bounds__(32) void flash_attn_wmma_kernel(
    const __bf16* __restrict__ Q, const __bf16* __restrict__ Km,
    const __bf16* __restrict__ V, const int* __restrict__ amask,
    __bf16* __restrict__ O, int B, int S, int H, float scale) {
  const int Dh = 128;
  const int HD = H * Dh;
  const int b = blockIdx.x / H;
  const int h = blockIdx.x % H;
  const int q0 = blockIdx.y * 16;
  const int lane = threadIdx.x;
  const int nrow = lane & 15;
  const int khalf = (lane >> 4) * 16;
  const int ncol = lane & 15;
  const int moff = (lane >> 4) * 8;

  __shared__ __align__(16) __bf16 sP[16][40];
  __shared__ __align__(16) __bf16 sV[32][136];

  // Q fragments: 4 chunks of 32 along Dh
  Frag qf[4];
  const __bf16* qbase = Q + (size_t)(b * S + q0 + nrow) * HD + (size_t)h * Dh;
#pragma unroll
  for (int c = 0; c < 4; ++c) {
    const __bf16* p = qbase + c * 32 + khalf;
    qf[c].u[0] = *(const uint4*)p;
    qf[c].u[1] = *(const uint4*)(p + 8);
  }

  const unsigned sv_dst = lds_addr32(&sV[lane][0]);

  v8f of[8] = {};
  float rmax[8], rsum[8];
#pragma unroll
  for (int e = 0; e < 8; ++e) {
    rmax[e] = -1e30f;
    rsum[e] = 0.0f;
  }

  const int klim = q0 + 16;  // exclusive causal bound
  for (int kb = 0; kb < klim; kb += 32) {
    // ---- scores: two 16x16 tiles covering k-cols [kb, kb+32)
    v8f sc[2] = {};
#pragma unroll
    for (int hlf = 0; hlf < 2; ++hlf) {
      int kr = kb + 16 * hlf + nrow;
      int krc = kr < S ? kr : S - 1;
      const __bf16* kbase = Km + (size_t)(b * S + krc) * HD + (size_t)h * Dh;
#pragma unroll
      for (int c = 0; c < 4; ++c) {
        Frag kf;
        const __bf16* p = kbase + c * 32 + khalf;
        kf.u[0] = *(const uint4*)p;
        kf.u[1] = *(const uint4*)(p + 8);
        sc[hlf] = wmma_bf16(qf[c].v, kf.v, sc[hlf]);
      }
    }

    // ---- kick off V tile (32 rows x 128 cols) into LDS asynchronously; the
    // softmax below overlaps the fill. Previous PV ds-reads already completed
    // (their DScnt was drained before the WMMAs consuming them).
    {
      int kr = kb + lane;
      int krc = kr < S ? kr : S - 1;
      const __bf16* vrow = V + (size_t)(b * S + krc) * HD + (size_t)h * Dh;
#pragma unroll
      for (int ts = 0; ts < 16; ++ts)
        async_load_b128(sv_dst + ts * 16, vrow + ts * 8);
    }

    __syncthreads();  // order sP reuse vs previous iteration's reads

    // ---- online softmax + write P (bf16) into LDS
    float alpha[8];
#pragma unroll
    for (int e = 0; e < 8; ++e) {
      const int qrow = q0 + e + moff;
      int kpos0 = kb + ncol;
      int kpos1 = kb + 16 + ncol;
      int kcl0 = kpos0 < S ? kpos0 : S - 1;
      int kcl1 = kpos1 < S ? kpos1 : S - 1;
      bool ok0 = (kpos0 <= qrow) && (amask[(size_t)b * S + kcl0] != 0);
      bool ok1 = (kpos1 <= qrow) && (amask[(size_t)b * S + kcl1] != 0);
      float s0 = sc[0][e] * scale + (ok0 ? 0.0f : -1e9f);
      float s1 = sc[1][e] * scale + (ok1 ? 0.0f : -1e9f);
      float mloc = fmaxf(s0, s1);
#pragma unroll
      for (int off = 1; off < 16; off <<= 1)
        mloc = fmaxf(mloc, __shfl_xor(mloc, off, 32));
      float mnew = fmaxf(rmax[e], mloc);
      float a = __expf(rmax[e] - mnew);
      float p0 = __expf(s0 - mnew);
      float p1 = __expf(s1 - mnew);
      float ps = p0 + p1;
#pragma unroll
      for (int off = 1; off < 16; off <<= 1) ps += __shfl_xor(ps, off, 32);
      rsum[e] = rsum[e] * a + ps;
      rmax[e] = mnew;
      alpha[e] = a;
      sP[e + moff][ncol] = (__bf16)p0;
      sP[e + moff][16 + ncol] = (__bf16)p1;
    }
#pragma unroll
    for (int j = 0; j < 8; ++j)
#pragma unroll
      for (int e = 0; e < 8; ++e) of[j][e] *= alpha[e];

    wait_asynccnt<0>();  // V tile is in LDS
    __syncthreads();

    // ---- PV: A = P (16x32), B = V chunk (32 x 16) per 16-wide d subtile
    Frag pf;
    {
      const __bf16* pp = &sP[nrow][khalf];
      pf.u[0] = *(const uint4*)pp;
      pf.u[1] = *(const uint4*)(pp + 8);
    }
#pragma unroll
    for (int j = 0; j < 8; ++j) {
      Frag vf;
#pragma unroll
      for (int i = 0; i < 16; ++i) vf.h[i] = sV[khalf + i][16 * j + nrow];
      of[j] = wmma_bf16(pf.v, vf.v, of[j]);
    }
  }

  // ---- normalize and write ctx (bf16)
#pragma unroll
  for (int j = 0; j < 8; ++j)
#pragma unroll
    for (int e = 0; e < 8; ++e) {
      int m = q0 + e + moff;
      int d = 16 * j + ncol;
      float val = of[j][e] / rsum[e];
      O[(size_t)(b * S + m) * HD + (size_t)h * Dh + d] = (__bf16)val;
    }
}

// ---------------------------------------------------------------------------
// Host orchestration
// ---------------------------------------------------------------------------
extern "C" void kernel_launch(void* const* d_in, const int* in_sizes, int n_in,
                              void* d_out, int out_size, void* d_ws,
                              size_t ws_size, hipStream_t stream) {
  (void)in_sizes; (void)n_in; (void)out_size; (void)ws_size;
  constexpr int L = 2, Bb = 2, S = 1024, D = 2048, F = 5504, R = 8, H = 16;
  constexpr int M = Bb * S;

  const float* inputs = (const float*)d_in[0];
  const int* amask = (const int*)d_in[1];
  const float* wq = (const float*)d_in[2];
  const float* aq = (const float*)d_in[3];
  const float* bq = (const float*)d_in[4];
  const float* wk = (const float*)d_in[5];
  const float* ak = (const float*)d_in[6];
  const float* bk = (const float*)d_in[7];
  const float* wv = (const float*)d_in[8];
  const float* av = (const float*)d_in[9];
  const float* bv = (const float*)d_in[10];
  const float* wo = (const float*)d_in[11];
  const float* ao = (const float*)d_in[12];
  const float* bo = (const float*)d_in[13];
  const float* ln1 = (const float*)d_in[14];
  const float* wg = (const float*)d_in[15];
  const float* wu = (const float*)d_in[16];
  const float* wd = (const float*)d_in[17];
  const float* ln2 = (const float*)d_in[18];
  const float* norm_f = (const float*)d_in[19];

  char* p = (char*)d_ws;
  auto carve = [&](size_t bytes) -> char* {
    char* r = p;
    p += (bytes + 255) & ~(size_t)255;
    return r;
  };
  float* x = (float*)carve((size_t)M * D * 4);
  __bf16* h_bf = (__bf16*)carve((size_t)M * D * 2);
  __bf16* wdeq = (__bf16*)carve((size_t)F * D * 2);
  __bf16* q_bf = (__bf16*)carve((size_t)M * D * 2);
  __bf16* k_bf = (__bf16*)carve((size_t)M * D * 2);
  __bf16* v_bf = (__bf16*)carve((size_t)M * D * 2);
  __bf16* c_bf = (__bf16*)carve((size_t)M * D * 2);
  float* yf = (float*)carve((size_t)M * F * 4);
  __bf16* g_bf = (__bf16*)carve((size_t)M * F * 2);
  float* tlr = (float*)carve((size_t)M * R * 4);

  hipMemcpyAsync(x, inputs, (size_t)M * D * sizeof(float),
                 hipMemcpyDeviceToDevice, stream);

  const float attn_scale = 0.08838834764831845f;  // 1/sqrt(128)

  auto deq = [&](const float* w, size_t numel) {
    int nb = (int)(numel / 64);
    nf4_dequant_kernel<<<dim3((nb + 255) / 256), dim3(256), 0, stream>>>(w, wdeq,
                                                                         nb);
  };
  auto gemm = [&](const __bf16* Ai, float* Ci, int N, int K) {
    gemm_bf16_wmma_kernel<<<dim3(M / 128, N / 128), dim3(256), 0, stream>>>(
        Ai, wdeq, Ci, M, N, K);
  };
  auto lora = [&](const __bf16* hin, const float* amat, const float* bmat,
                  float* y, int N) {
    lora_t_kernel<<<dim3(M * R / 8), dim3(256), 0, stream>>>(hin, amat, tlr, M,
                                                             R, D);
    size_t tot = (size_t)M * N;
    lora_add_kernel<<<dim3((unsigned)((tot + 255) / 256)), dim3(256), 0,
                      stream>>>(y, tlr, bmat, M, N, R, 2.0f);
  };
  auto ew_grid = [](size_t n) { return dim3((unsigned)((n + 255) / 256)); };

  for (int l = 0; l < L; ++l) {
    const float* wq_l = wq + (size_t)l * D * D;
    const float* aq_l = aq + (size_t)l * R * D;
    const float* bq_l = bq + (size_t)l * D * R;
    const float* wk_l = wk + (size_t)l * D * D;
    const float* ak_l = ak + (size_t)l * R * D;
    const float* bk_l = bk + (size_t)l * D * R;
    const float* wv_l = wv + (size_t)l * D * D;
    const float* av_l = av + (size_t)l * R * D;
    const float* bv_l = bv + (size_t)l * D * R;
    const float* wo_l = wo + (size_t)l * D * D;
    const float* ao_l = ao + (size_t)l * R * D;
    const float* bo_l = bo + (size_t)l * D * R;
    const float* ln1_l = ln1 + (size_t)l * D;
    const float* wg_l = wg + (size_t)l * F * D;
    const float* wu_l = wu + (size_t)l * F * D;
    const float* wd_l = wd + (size_t)l * D * F;
    const float* ln2_l = ln2 + (size_t)l * D;

    // ---- attention block
    rmsnorm_kernel<__bf16><<<dim3(M), dim3(256), 0, stream>>>(x, ln1_l, h_bf, D);

    // Q
    deq(wq_l, (size_t)D * D);
    gemm(h_bf, yf, D, D);
    lora(h_bf, aq_l, bq_l, yf, D);
    rope_to_bf16_kernel<<<ew_grid((size_t)M * H * 64), dim3(256), 0, stream>>>(
        yf, q_bf, M, S, H);
    // K
    deq(wk_l, (size_t)D * D);
    gemm(h_bf, yf, D, D);
    lora(h_bf, ak_l, bk_l, yf, D);
    rope_to_bf16_kernel<<<ew_grid((size_t)M * H * 64), dim3(256), 0, stream>>>(
        yf, k_bf, M, S, H);
    // V (no rope)
    deq(wv_l, (size_t)D * D);
    gemm(h_bf, yf, D, D);
    lora(h_bf, av_l, bv_l, yf, D);
    tobf16_kernel<<<ew_grid((size_t)M * D), dim3(256), 0, stream>>>(
        yf, v_bf, (size_t)M * D);

    flash_attn_wmma_kernel<<<dim3(Bb * H, S / 16), dim3(32), 0, stream>>>(
        q_bf, k_bf, v_bf, amask, c_bf, Bb, S, H, attn_scale);

    // O projection + residual
    deq(wo_l, (size_t)D * D);
    gemm(c_bf, yf, D, D);
    lora(c_bf, ao_l, bo_l, yf, D);
    add_residual_kernel<<<ew_grid((size_t)M * D), dim3(256), 0, stream>>>(
        x, yf, (size_t)M * D);

    // ---- MLP block
    rmsnorm_kernel<__bf16><<<dim3(M), dim3(256), 0, stream>>>(x, ln2_l, h_bf, D);

    deq(wg_l, (size_t)F * D);
    gemm(h_bf, yf, F, D);
    silu_bf16_kernel<<<ew_grid((size_t)M * F), dim3(256), 0, stream>>>(
        yf, g_bf, (size_t)M * F);

    deq(wu_l, (size_t)F * D);
    gemm(h_bf, yf, F, D);
    mul_bf16_kernel<<<ew_grid((size_t)M * F), dim3(256), 0, stream>>>(
        yf, g_bf, (size_t)M * F);

    deq(wd_l, (size_t)D * F);
    gemm(g_bf, yf, D, F);
    add_residual_kernel<<<ew_grid((size_t)M * D), dim3(256), 0, stream>>>(
        x, yf, (size_t)M * D);
  }

  rmsnorm_kernel<float><<<dim3(M), dim3(256), 0, stream>>>(x, norm_f,
                                                           (float*)d_out, D);
}
